// SpanEndpoints_5995774345595
// MI455X (gfx1250) — matrix operand order repaired and checked by reading, not compiled, not run on hardware
//
#include <hip/hip_runtime.h>

// Problem constants (from reference): B=8, L=512, D=1024, W=16
constexpr int kB     = 8;
constexpr int kL     = 512;
constexpr int kD     = 1024;            // output features; also half of K
constexpr int kK     = 2048;            // inner dim = 2*D
constexpr int kNspan = kL * 16;         // spans per batch = 8192
constexpr int kHElems = kB * kL * kD;   // 4,194,304
constexpr int kWElems = kD * kK;        // 2,097,152
constexpr int kSpans  = kB * kNspan;    // 65,536

// Tiling
constexpr int BM  = 256;
constexpr int BN  = 128;
constexpr int BK  = 64;                 // two wmma k-substeps per staging round
constexpr int LDP = 72;                 // LDS pitch in bf16 elems: 64 data + 8 pad (144B)

// Workspace layout (bytes): bf16 h copy | bf16 W1 copy | clipped span row offsets
constexpr size_t WS_HB   = 0;
constexpr size_t WS_WB   = WS_HB + (size_t)kHElems * sizeof(unsigned short); // +8 MB
constexpr size_t WS_SOFF = WS_WB + (size_t)kWElems * sizeof(unsigned short); // +4 MB

typedef __attribute__((ext_vector_type(16))) __bf16         v16bf;
typedef __attribute__((ext_vector_type(16))) unsigned short v16us;
typedef __attribute__((ext_vector_type(8)))  unsigned short v8us;
typedef __attribute__((ext_vector_type(8)))  float          v8f;

union FragU {
    v8us  h[2];
    v16us u;
    v16bf b;
};

__device__ __forceinline__ unsigned short f2bf(float x) {
    unsigned int u = __float_as_uint(x);
    u += 0x7FFFu + ((u >> 16) & 1u);     // round-to-nearest-even
    return (unsigned short)(u >> 16);
}

// ---------------------------------------------------------------------------
// Prep: fp32 -> bf16 copies of h and W1; clip int64 span endpoints to int32
// flat row offsets into h. One-shot, pure bandwidth (~25 MB).
// ---------------------------------------------------------------------------
__global__ __launch_bounds__(256)
void span_prep_kernel(const float* __restrict__ h,
                      const long long* __restrict__ span_idx,
                      const float* __restrict__ W1,
                      unsigned short* __restrict__ hb,
                      unsigned short* __restrict__ wb,
                      int* __restrict__ soff) {
    const int idx = blockIdx.x * 256 + threadIdx.x;
    constexpr int nH4 = kHElems / 4;
    constexpr int nW4 = kWElems / 4;

    if (idx < nH4) {
        const float4 v = ((const float4*)h)[idx];
        ((ushort4*)hb)[idx] = make_ushort4(f2bf(v.x), f2bf(v.y), f2bf(v.z), f2bf(v.w));
    } else if (idx < nH4 + nW4) {
        const int i = idx - nH4;
        const float4 v = ((const float4*)W1)[i];
        ((ushort4*)wb)[i] = make_ushort4(f2bf(v.x), f2bf(v.y), f2bf(v.z), f2bf(v.w));
    } else if (idx < nH4 + nW4 + kSpans) {
        const int i = idx - nH4 - nW4;
        const int b = i >> 13;
        const long long* sp = span_idx + (size_t)i * 2;
        long long s = sp[0];
        long long e = sp[1];
        s = s < 0 ? 0 : (s > kL - 1 ? (long long)(kL - 1) : s);
        e = e < 0 ? 0 : (e > kL - 1 ? (long long)(kL - 1) : e);
        soff[2 * i]     = (b * kL + (int)s) * kD;
        soff[2 * i + 1] = (b * kL + (int)e) * kD;
    }
}

// ---------------------------------------------------------------------------
// Gathered GEMM via bf16 WMMA, async global->LDS staging, double-buffered,
// single barrier per K-step. 256x128 tile / 8 waves (4x2), 64x64 per wave:
// 16 ds_load_b128 feed 16 v_wmma per substep (1.0 LDS op per WMMA).
// ---------------------------------------------------------------------------
__global__ __launch_bounds__(256)
void span_ffn_wmma_kernel(const unsigned short* __restrict__ hb,
                          const unsigned short* __restrict__ wb,
                          const int* __restrict__ soff,
                          const float* __restrict__ b1,
                          float* __restrict__ out) {
    __shared__ int            sRowS[BM];
    __shared__ int            sRowE[BM];
    __shared__ unsigned short Alds[2][BM * LDP];   // 2 x 36 KB
    __shared__ unsigned short Blds[2][BN * LDP];   // 2 x 18 KB

    const int tid = threadIdx.x;
    const int b   = blockIdx.z;
    const int bm  = blockIdx.y * BM;
    const int bn  = blockIdx.x * BN;

    {
        const int g = 2 * (b * kNspan + bm + tid);
        sRowS[tid] = soff[g];
        sRowE[tid] = soff[g + 1];
    }
    __syncthreads();

    // A staging: BM rows x 8 16B-chunks = 2048 chunks, 8 per thread.
    int      aRow[8], aC8[8];
    unsigned aVoffS[8], aVoffE[8], aLds[8];
    #pragma unroll
    for (int i = 0; i < 8; i++) {
        const int idx = tid + i * 256;
        aRow[i] = idx >> 3;
        aC8[i]  = (idx & 7) * 8;
        aVoffS[i] = (unsigned)((sRowS[aRow[i]] + aC8[i]) * 2);
        aVoffE[i] = (unsigned)((sRowE[aRow[i]] + aC8[i]) * 2);
        aLds[i]   = (unsigned)(uintptr_t)&Alds[0][aRow[i] * LDP + aC8[i]];
    }
    // B staging: BN rows x 8 chunks = 1024 chunks, 4 per thread.
    unsigned bVoff[4], bLds[4];
    #pragma unroll
    for (int i = 0; i < 4; i++) {
        const int idx = tid + i * 256;
        const int row = idx >> 3;
        const int c8  = (idx & 7) * 8;
        bVoff[i] = (unsigned)(((bn + row) * kK + c8) * 2);
        bLds[i]  = (unsigned)(uintptr_t)&Blds[0][row * LDP + c8];
    }

    constexpr unsigned ABUFB = (unsigned)(BM * LDP * 2);
    constexpr unsigned BBUFB = (unsigned)(BN * LDP * 2);

    auto issueAsync = [&](int kk, int buf) {
        const bool     useEnd = (kk >= kD);
        const unsigned kbA    = (unsigned)((kk & (kD - 1)) * 2);
        const unsigned kbB    = (unsigned)(kk * 2);
        const unsigned aBuf   = buf ? ABUFB : 0u;
        const unsigned bBuf   = buf ? BBUFB : 0u;
        #pragma unroll
        for (int i = 0; i < 8; i++) {
            const unsigned av = (useEnd ? aVoffE[i] : aVoffS[i]) + kbA;
            const unsigned al = aLds[i] + aBuf;
            asm volatile("global_load_async_to_lds_b128 %0, %1, %2"
                         :: "v"(al), "v"(av), "s"(hb) : "memory");
        }
        #pragma unroll
        for (int i = 0; i < 4; i++) {
            const unsigned bv = bVoff[i] + kbB;
            const unsigned bl = bLds[i] + bBuf;
            asm volatile("global_load_async_to_lds_b128 %0, %1, %2"
                         :: "v"(bl), "v"(bv), "s"(wb) : "memory");
        }
    };

    const int wave = tid >> 5;
    const int lane = tid & 31;
    const int wm   = (wave & 3) * 64;    // 4 waves down M
    const int wn   = (wave >> 2) * 64;   // 2 waves across N
    const int l15  = lane & 15;
    const int hi   = lane >> 4;

    v8f acc[4][4] = {};

    issueAsync(0, 0);
    int buf = 0;
    for (int kk = 0; kk < kK; kk += BK, buf ^= 1) {
        asm volatile("s_wait_asynccnt 0x0" ::: "memory");
        __syncthreads();                           // publishes buf; frees buf^1
        if (kk + BK < kK) issueAsync(kk + BK, buf ^ 1);

        const unsigned short* Ab = Alds[buf];
        const unsigned short* Bb = Blds[buf];
        #pragma unroll
        for (int s = 0; s < BK; s += 32) {
            // A frags (16x32 bf16): per-lane chunks at K = {hi*8, 16+hi*8}
            FragU afr[4];
            #pragma unroll
            for (int mf = 0; mf < 4; mf++) {
                const unsigned short* p = &Ab[(wm + mf * 16 + l15) * LDP + s + hi * 8];
                afr[mf].h[0] = *(const v8us*)(p);
                afr[mf].h[1] = *(const v8us*)(p + 16);
            }
            // B frags (32x16): per-lane chunks at K = {hi*8, 16+hi*8}
            #pragma unroll
            for (int nf = 0; nf < 4; nf++) {
                FragU bfr;
                const unsigned short* p = &Bb[(wn + nf * 16 + l15) * LDP + s + hi * 8];
                bfr.h[0] = *(const v8us*)(p);
                bfr.h[1] = *(const v8us*)(p + 16);

                #pragma unroll
                for (int mf = 0; mf < 4; mf++) {
                    acc[mf][nf] = __builtin_amdgcn_wmma_f32_16x16x32_bf16(
                        false, afr[mf].b, false, bfr.b,
                        (short)0, acc[mf][nf], false, false);
                }
            }
        }
    }

    // Epilogue: bias + ReLU. C/D layout: lanes 0-15 N=lane,M=r; 16-31 N=l15,M=r+8.
    #pragma unroll
    for (int mf = 0; mf < 4; mf++) {
        const int rbase = bm + wm + mf * 16 + hi * 8;
        #pragma unroll
        for (int nf = 0; nf < 4; nf++) {
            const int col  = bn + wn + nf * 16 + l15;
            const float bv = b1[col];
            #pragma unroll
            for (int r = 0; r < 8; r++) {
                float v = acc[mf][nf][r] + bv;
                v = v > 0.0f ? v : 0.0f;
                out[((size_t)b * kNspan + rbase + r) * kD + col] = v;
            }
        }
    }
}

extern "C" void kernel_launch(void* const* d_in, const int* in_sizes, int n_in,
                              void* d_out, int out_size, void* d_ws, size_t ws_size,
                              hipStream_t stream) {
    const float*     h        = (const float*)d_in[0];
    const long long* span_idx = (const long long*)d_in[1];
    const float*     W1       = (const float*)d_in[2];
    const float*     b1       = (const float*)d_in[3];
    float*           out      = (float*)d_out;

    unsigned short* hb   = (unsigned short*)((char*)d_ws + WS_HB);
    unsigned short* wb   = (unsigned short*)((char*)d_ws + WS_WB);
    int*            soff = (int*)((char*)d_ws + WS_SOFF);

    constexpr int prepTasks  = kHElems / 4 + kWElems / 4 + kSpans;
    constexpr int prepBlocks = (prepTasks + 255) / 256;
    span_prep_kernel<<<prepBlocks, 256, 0, stream>>>(h, span_idx, W1, hb, wb, soff);

    dim3 grid(kD / BN,       // 8  N-tiles
              kNspan / BM,   // 32 M-tiles per batch
              kB);           // 8  batches
    span_ffn_wmma_kernel<<<grid, 256, 0, stream>>>(hb, wb, soff, b1, out);
}
